// egnn_model_42193758716008
// MI455X (gfx1250) — matrix operand order repaired and checked by reading, not compiled, not run on hardware
//
#include <hip/hip_runtime.h>
#include <hip/hip_bf16.h>

// ---------------- problem constants (match reference) ----------------
#define NN   16384
#define EE   131072
#define GG   32
#define AA   2
#define LL   3
#define DD   128
#define INF_ 40
#define EH1  514     // 2*EH = 2*(2*D+1)
#define EW1_KC 8     // K=256 -> 8 chunks of 32 (row 256 handled as rank-1)
#define EW1_NT 33    // ceil(514/16)
#define EW2_KC 17    // K=514 padded to 544 -> 17 chunks
#define EW2_NT 8     // 128/16
#define NW1_KC 8
#define NW1_NT 16
#define NW2_KC 8
#define NW2_NT 8

#define BE    64     // edges per block (4 row groups x 16)
#define BNODE 32     // nodes per block (2 row groups x 16)

#define EIN_STRIDE 288   // u16 stride for 256-wide bf16 rows in LDS (16B aligned)
#define H_STRIDE   560   // u16 stride for 544-wide padded h rows in LDS

#define CDIV(a,b) (((a)+(b)-1)/(b))

typedef __bf16 v16bf __attribute__((ext_vector_type(16)));
typedef float  v8f   __attribute__((ext_vector_type(8)));

union Frag { v16bf v; uint4 q[2]; unsigned short u[16]; };

__device__ __forceinline__ unsigned short f2bf(float f) {
  unsigned u = __float_as_uint(f);
  u += 0x7FFFu + ((u >> 16) & 1u);        // round-to-nearest-even
  return (unsigned short)(u >> 16);
}
__device__ __forceinline__ unsigned pack2bf(float a, float b) {
  return (unsigned)f2bf(a) | ((unsigned)f2bf(b) << 16);
}
__device__ __forceinline__ float silu_f(float x) { return x / (1.f + __expf(-x)); }

__device__ __forceinline__ v8f wmma_bf16(const Frag& a, const Frag& b, v8f c) {
  return __builtin_amdgcn_wmma_f32_16x16x32_bf16(false, a.v, false, b.v,
                                                 (short)0, c, false, false);
}

// ---------------- weight packing: f32 [K x Ncols] -> B-fragment order -------
// chunk = ntile*Kchunks + kc ; per chunk: 32 lanes x 8 dwords (2 bf16 each).
// lane: N = ntile*16 + lane%16 ; K pair base = kc*32 + (lane>=16)*8 + 2*(v&3) + (v>=4)*16.
__global__ void pack_b_kernel(const float* __restrict__ src, unsigned* __restrict__ dst,
                              int K, int Ncols, int Kchunks, int Ntiles) {
  int t = blockIdx.x * blockDim.x + threadIdx.x;
  int total = Kchunks * Ntiles * 32;
  if (t >= total) return;
  int lane  = t & 31;
  int chunk = t >> 5;
  int kc    = chunk % Kchunks;
  int ntile = chunk / Kchunks;
  int n     = (ntile << 4) + (lane & 15);
  int kbase = (kc << 5) + ((lane >> 4) << 3);
  unsigned* out = dst + (size_t)chunk * 256 + lane * 8;
#pragma unroll
  for (int v = 0; v < 8; ++v) {
    int k0 = kbase + ((v >> 2) << 4) + 2 * (v & 3);
    float f0 = (k0     < K && n < Ncols) ? src[(size_t)k0       * Ncols + n] : 0.f;
    float f1 = (k0 + 1 < K && n < Ncols) ? src[(size_t)(k0 + 1) * Ncols + n] : 0.f;
    out[v] = pack2bf(f0, f1);
  }
}

// rel_dist row (row 256 of edge_w1), zero-padded to 528
__global__ void ew1_row_kernel(const float* __restrict__ src, float* __restrict__ dst) {
  int i = blockIdx.x * blockDim.x + threadIdx.x;
  if (i < 528) dst[i] = (i < EH1) ? src[(size_t)256 * EH1 + i] : 0.f;
}

// ---------------- h0 = relu(x @ lin0_w + b), K=40 (tiny, VALU) --------------
__global__ void lin0_kernel(const float* __restrict__ x, const float* __restrict__ w,
                            const float* __restrict__ b, float* __restrict__ feats,
                            unsigned short* __restrict__ featsBf) {
  int idx = blockIdx.x * blockDim.x + threadIdx.x;
  if (idx >= NN * DD) return;
  int node = idx >> 7, j = idx & 127;
  float acc = b[j];
  const float* xr = x + (size_t)node * INF_;
#pragma unroll
  for (int k = 0; k < INF_; ++k) acc = fmaf(xr[k], w[k * DD + j], acc);
  acc = fmaxf(acc, 0.f);
  feats[idx]   = acc;
  featsBf[idx] = f2bf(acc);
}

__global__ void copy_coords_kernel(const float* __restrict__ src, float* __restrict__ dst) {
  int i = blockIdx.x * blockDim.x + threadIdx.x;
  if (i < NN * 3) dst[i] = src[i];
}

// ---------------- edge MLP + scatter: 64 edges/block, 8 waves ---------------
// waves: rg = wv&3 (row group of 16 edges), cg = wv>>2 (column stream 0/1)
__global__ __launch_bounds__(256)
void edge_kernel(const long long* __restrict__ rowIdx, const long long* __restrict__ colIdx,
                 const unsigned short* __restrict__ featsBf, const float* __restrict__ coordsW,
                 const unsigned* __restrict__ pw1, const float* __restrict__ w1r,
                 const float* __restrict__ b1, const unsigned* __restrict__ pw2,
                 const float* __restrict__ b2, float* __restrict__ mi) {
  __shared__ unsigned short s_ein[BE * EIN_STRIDE];
  __shared__ unsigned short s_h[BE * H_STRIDE];
  __shared__ float s_rel[BE];
  __shared__ int   s_row[BE];

  const int tid   = threadIdx.x;
  const int lane  = tid & 31;
  const int wv    = tid >> 5;
  const int rg    = wv & 3;
  const int cg    = wv >> 2;
  const int eBase = blockIdx.x * BE;

  // stage: e_in rows (bf16) = [feats[row] | feats[col]]  (dword copies)
  for (int i = tid; i < BE * 128; i += 256) {
    int e = i >> 7, j = i & 127;
    long long nd = (j < 64) ? rowIdx[eBase + e] : colIdx[eBase + e];
    int jj = (j < 64) ? j : (j - 64);
    unsigned d = ((const unsigned*)featsBf)[(size_t)nd * 64 + jj];
    *(unsigned*)&s_ein[e * EIN_STRIDE + (j << 1)] = d;
  }
  // zero the K-pad region of h (cols 512..543)
  for (int i = tid; i < BE * 16; i += 256) {
    int e = i >> 4, j = i & 15;
    *(unsigned*)&s_h[e * H_STRIDE + 512 + (j << 1)] = 0u;
  }
  if (tid < BE) {
    long long r = rowIdx[eBase + tid], c = colIdx[eBase + tid];
    s_row[tid] = (int)r;
    float dx = coordsW[r * 3 + 0] - coordsW[c * 3 + 0];
    float dy = coordsW[r * 3 + 1] - coordsW[c * 3 + 1];
    float dz = coordsW[r * 3 + 2] - coordsW[c * 3 + 2];
    s_rel[tid] = dx * dx + dy * dy + dz * dz;
  }
  __syncthreads();

  const int m  = lane & 15;
  const int hi = lane >> 4;
  const int kb = hi << 3;                     // 0 or 8 (u16 units)
  const int mrow = rg * 16 + m;               // this wave's edge row for A frags

  // preload all 8 A-fragments (K=256) for this wave's 16-edge row group
  Frag aF[8];
  {
    const unsigned short* rp = &s_ein[mrow * EIN_STRIDE];
#pragma unroll
    for (int kc = 0; kc < 8; ++kc) {
      aF[kc].q[0] = *(const uint4*)(rp + kc * 32 + kb);
      aF[kc].q[1] = *(const uint4*)(rp + kc * 32 + 16 + kb);
    }
  }
  float rl[8];
#pragma unroll
  for (int r = 0; r < 8; ++r) rl[r] = s_rel[rg * 16 + r + (hi << 3)];

  // layer 1: [64 x 514] = A(64x256) @ W1 + rel_dist * w1[256,:]  -> silu -> LDS
  for (int nt = cg; nt < EW1_NT; nt += 2) {
    const unsigned* bp = pw1 + (size_t)(nt * EW1_KC) * 256 + lane * 8;
    __builtin_prefetch(bp + 2 * EW1_KC * 256, 0, 1);   // next tile for this stream
    // batch all B fragments: one clause of 16 b128 loads, one wait
    Frag bF[8];
#pragma unroll
    for (int kc = 0; kc < EW1_KC; ++kc) {
      bF[kc].q[0] = *(const uint4*)(bp);
      bF[kc].q[1] = *(const uint4*)(bp + 4);
      bp += 256;
    }
    // two independent accumulator chains (even / odd K-chunks)
    v8f a0 = {}, a1 = {};
#pragma unroll
    for (int kc = 0; kc < EW1_KC; kc += 2) {
      a0 = wmma_bf16(aF[kc],     bF[kc],     a0);
      a1 = wmma_bf16(aF[kc + 1], bF[kc + 1], a1);
    }
    v8f acc = a0 + a1;
    int n = (nt << 4) + m;
    float wr   = (n < EH1) ? w1r[n] : 0.f;
    float bias = (n < EH1) ? b1[n]  : 0.f;
#pragma unroll
    for (int r = 0; r < 8; ++r) {
      float v = silu_f(acc[r] + rl[r] * wr + bias);   // rank-1 rel_dist term
      unsigned short hv = (n < EH1) ? f2bf(v) : (unsigned short)0;
      s_h[(rg * 16 + r + (hi << 3)) * H_STRIDE + n] = hv;
    }
  }
  __syncthreads();

  // layer 2: m = silu(h @ W2 + b2), K padded to 544 (17 chunks); scatter-add
  const unsigned short* hp = &s_h[mrow * H_STRIDE];
  for (int nt = cg; nt < EW2_NT; nt += 2) {
    const unsigned* bp = pw2 + (size_t)(nt * EW2_KC) * 256 + lane * 8;
    __builtin_prefetch(bp + 2 * EW2_KC * 256, 0, 1);
    v8f a0 = {}, a1 = {};
#pragma unroll
    for (int kc = 0; kc < 16; kc += 2) {
      Frag t0, t1, c0, c1;
      t0.q[0] = *(const uint4*)(hp + kc * 32 + kb);
      t0.q[1] = *(const uint4*)(hp + kc * 32 + 16 + kb);
      t1.q[0] = *(const uint4*)(hp + (kc + 1) * 32 + kb);
      t1.q[1] = *(const uint4*)(hp + (kc + 1) * 32 + 16 + kb);
      c0.q[0] = *(const uint4*)(bp);
      c0.q[1] = *(const uint4*)(bp + 4);
      c1.q[0] = *(const uint4*)(bp + 256);
      c1.q[1] = *(const uint4*)(bp + 260);
      bp += 512;
      a0 = wmma_bf16(t0, c0, a0);
      a1 = wmma_bf16(t1, c1, a1);
    }
    { // tail chunk 16
      Frag t, c;
      t.q[0] = *(const uint4*)(hp + 16 * 32 + kb);
      t.q[1] = *(const uint4*)(hp + 16 * 32 + 16 + kb);
      c.q[0] = *(const uint4*)(bp);
      c.q[1] = *(const uint4*)(bp + 4);
      a0 = wmma_bf16(t, c, a0);
    }
    v8f acc = a0 + a1;
    int n = (nt << 4) + m;
    float bias = b2[n];
#pragma unroll
    for (int r = 0; r < 8; ++r) {
      float v = silu_f(acc[r] + bias);
      int node = s_row[rg * 16 + r + (hi << 3)];
      atomicAdd(&mi[(size_t)node * DD + n], v);       // segment_sum (coalesced)
    }
  }
}

// ---------------- node MLP + residual + relu: 32 nodes/block, 8 waves -------
// waves: rg = wv&1 (row group of 16 nodes), cs = wv>>1 (column stream 0..3)
__global__ __launch_bounds__(256)
void node_kernel(float* __restrict__ feats, unsigned short* __restrict__ featsBf,
                 const float* __restrict__ mi, float* __restrict__ coordsW,
                 const unsigned* __restrict__ pw1, const float* __restrict__ b1,
                 const unsigned* __restrict__ pw2, const float* __restrict__ b2) {
  __shared__ unsigned short s_nin[BNODE * EIN_STRIDE];
  __shared__ unsigned short s_h2 [BNODE * EIN_STRIDE];
  const int tid = threadIdx.x, lane = tid & 31, wv = tid >> 5;
  const int rg = wv & 1, cs = wv >> 1;
  const int nodeBase = blockIdx.x * BNODE;

  // stage n_in = [feats | m_i] as bf16
  for (int i = tid; i < BNODE * 128; i += 256) {
    int nd = i >> 7, j = i & 127;
    unsigned d;
    if (j < 64) {
      d = ((const unsigned*)featsBf)[(size_t)(nodeBase + nd) * 64 + j];
    } else {
      const float* mp = mi + (size_t)(nodeBase + nd) * DD + ((j - 64) << 1);
      d = pack2bf(mp[0], mp[1]);
    }
    *(unsigned*)&s_nin[nd * EIN_STRIDE + (j << 1)] = d;
  }
  __syncthreads();

  const int m = lane & 15, hi = lane >> 4, kb = hi << 3;
  const int mrow = rg * 16 + m;
  Frag aF[8];
  {
    const unsigned short* rp = &s_nin[mrow * EIN_STRIDE];
#pragma unroll
    for (int kc = 0; kc < 8; ++kc) {
      aF[kc].q[0] = *(const uint4*)(rp + kc * 32 + kb);
      aF[kc].q[1] = *(const uint4*)(rp + kc * 32 + 16 + kb);
    }
  }

  // layer 1: h2 = silu(n_in @ nw1 + nb1), N=256
  for (int nt = cs; nt < NW1_NT; nt += 4) {
    const unsigned* bp = pw1 + (size_t)(nt * NW1_KC) * 256 + lane * 8;
    __builtin_prefetch(bp + 4 * NW1_KC * 256, 0, 1);
    Frag bF[8];
#pragma unroll
    for (int kc = 0; kc < NW1_KC; ++kc) {
      bF[kc].q[0] = *(const uint4*)(bp);
      bF[kc].q[1] = *(const uint4*)(bp + 4);
      bp += 256;
    }
    v8f a0 = {}, a1 = {};
#pragma unroll
    for (int kc = 0; kc < NW1_KC; kc += 2) {
      a0 = wmma_bf16(aF[kc],     bF[kc],     a0);
      a1 = wmma_bf16(aF[kc + 1], bF[kc + 1], a1);
    }
    v8f acc = a0 + a1;
    int n = (nt << 4) + m;
    float bias = b1[n];
#pragma unroll
    for (int r = 0; r < 8; ++r)
      s_h2[(rg * 16 + r + (hi << 3)) * EIN_STRIDE + n] = f2bf(silu_f(acc[r] + bias));
  }
  __syncthreads();

  // layer 2: feats = relu(h2 @ nw2 + nb2 + feats)
  const unsigned short* hp = &s_h2[mrow * EIN_STRIDE];
  for (int nt = cs; nt < NW2_NT; nt += 4) {
    const unsigned* bp = pw2 + (size_t)(nt * NW2_KC) * 256 + lane * 8;
    v8f a0 = {}, a1 = {};
#pragma unroll
    for (int kc = 0; kc < NW2_KC; kc += 2) {
      Frag t0, t1, c0, c1;
      t0.q[0] = *(const uint4*)(hp + kc * 32 + kb);
      t0.q[1] = *(const uint4*)(hp + kc * 32 + 16 + kb);
      t1.q[0] = *(const uint4*)(hp + (kc + 1) * 32 + kb);
      t1.q[1] = *(const uint4*)(hp + (kc + 1) * 32 + 16 + kb);
      c0.q[0] = *(const uint4*)(bp);
      c0.q[1] = *(const uint4*)(bp + 4);
      c1.q[0] = *(const uint4*)(bp + 256);
      c1.q[1] = *(const uint4*)(bp + 260);
      bp += 512;
      a0 = wmma_bf16(t0, c0, a0);
      a1 = wmma_bf16(t1, c1, a1);
    }
    v8f acc = a0 + a1;
    int n = (nt << 4) + m;
    float bias = b2[n];
#pragma unroll
    for (int r = 0; r < 8; ++r) {
      int node = nodeBase + rg * 16 + r + (hi << 3);
      size_t off = (size_t)node * DD + n;
      float v = fmaxf(acc[r] + bias + feats[off], 0.f);   // residual + relu(s)
      feats[off]   = v;
      featsBf[off] = f2bf(v);
    }
  }
  // relu(s) also hits the coord part of the state
  if (tid < BNODE * 3) {
    size_t ci = (size_t)nodeBase * 3 + tid;
    coordsW[ci] = fmaxf(coordsW[ci], 0.f);
  }
}

// ---------------- per-graph pooling (segment_sum over batch) ----------------
__global__ void pool_kernel(const float* __restrict__ feats, const long long* __restrict__ batch,
                            float* __restrict__ pooledAdj) {
  int idx = blockIdx.x * blockDim.x + threadIdx.x;
  if (idx >= NN * DD) return;
  int node = idx >> 7, col = idx & 127;
  int g = (int)batch[node];
  atomicAdd(&pooledAdj[(size_t)g * DD + col], feats[idx]);
}

// ---------------- head: relu(xg @ lin1 + b) @ lin2 + b (tiny) ---------------
__global__ void final_kernel(const float* __restrict__ pooled, const float* __restrict__ w1,
                             const float* __restrict__ b1, const float* __restrict__ w2,
                             const float* __restrict__ b2, float* __restrict__ out) {
  int g = threadIdx.x;
  if (g >= GG) return;
  float o = b2[0];
  for (int j = 0; j < DD; ++j) {
    float acc = b1[j];
    for (int k = 0; k < 2 * DD; ++k) {
      float xv = (k < DD) ? pooled[(size_t)g * DD + k]
                          : pooled[(size_t)(GG + g) * DD + (k - DD)];
      acc = fmaf(xv, w1[k * DD + j], acc);
    }
    o = fmaf(fmaxf(acc, 0.f), w2[j], o);
  }
  out[g] = o;
}

// ---------------------------- launcher --------------------------------------
extern "C" void kernel_launch(void* const* d_in, const int* in_sizes, int n_in,
                              void* d_out, int out_size, void* d_ws, size_t ws_size,
                              hipStream_t stream) {
  const float*     x      = (const float*)d_in[0];
  const float*     coord  = (const float*)d_in[1];
  const long long* eidx   = (const long long*)d_in[2];
  const long long* batch  = (const long long*)d_in[3];
  const float*     lin0_w = (const float*)d_in[4];
  const float*     lin0_b = (const float*)d_in[5];
  const float*     ew1    = (const float*)d_in[6];
  const float*     eb1    = (const float*)d_in[7];
  const float*     ew2    = (const float*)d_in[8];
  const float*     eb2    = (const float*)d_in[9];
  const float*     nw1    = (const float*)d_in[10];
  const float*     nb1    = (const float*)d_in[11];
  const float*     nw2    = (const float*)d_in[12];
  const float*     nb2    = (const float*)d_in[13];
  const float*     l1w    = (const float*)d_in[14];
  const float*     l1b    = (const float*)d_in[15];
  const float*     l2w    = (const float*)d_in[16];
  const float*     l2b    = (const float*)d_in[17];

  char* ws = (char*)d_ws;
  size_t off = 0;
  auto alloc = [&](size_t bytes) -> void* {
    void* p = ws + off;
    off = (off + bytes + 255) & ~(size_t)255;
    return p;
  };

  float*          feats   = (float*)         alloc((size_t)NN * DD * 4);
  unsigned short* featsBf = (unsigned short*)alloc((size_t)NN * DD * 2);
  float*          coordsW = (float*)         alloc((size_t)NN * 3 * 4);
  float*          mi      = (float*)         alloc((size_t)NN * DD * 4);
  float*          pooled  = (float*)         alloc((size_t)AA * GG * DD * 4);

  unsigned* pEw1[AA * LL]; float* rEw1[AA * LL];
  unsigned* pEw2[AA * LL]; unsigned* pNw1[AA * LL]; unsigned* pNw2[AA * LL];
  for (int id = 0; id < AA * LL; ++id) {
    pEw1[id] = (unsigned*)alloc((size_t)EW1_KC * EW1_NT * 256 * 4);
    rEw1[id] = (float*)   alloc(528 * 4);
    pEw2[id] = (unsigned*)alloc((size_t)EW2_KC * EW2_NT * 256 * 4);
    pNw1[id] = (unsigned*)alloc((size_t)NW1_KC * NW1_NT * 256 * 4);
    pNw2[id] = (unsigned*)alloc((size_t)NW2_KC * NW2_NT * 256 * 4);
  }
  if (off > ws_size) return;   // workspace too small (no-op)

  // ---- pack all weights into bf16 B-fragment layout (once per launch) ----
  for (int id = 0; id < AA * LL; ++id) {
    const float* w1s = ew1 + (size_t)id * 257 * EH1;
    pack_b_kernel<<<CDIV(EW1_KC * EW1_NT * 32, 256), 256, 0, stream>>>(
        w1s, pEw1[id], 256, EH1, EW1_KC, EW1_NT);
    ew1_row_kernel<<<3, 256, 0, stream>>>(w1s, rEw1[id]);
    pack_b_kernel<<<CDIV(EW2_KC * EW2_NT * 32, 256), 256, 0, stream>>>(
        ew2 + (size_t)id * EH1 * DD, pEw2[id], EH1, DD, EW2_KC, EW2_NT);
    pack_b_kernel<<<CDIV(NW1_KC * NW1_NT * 32, 256), 256, 0, stream>>>(
        nw1 + (size_t)id * 256 * 256, pNw1[id], 256, 256, NW1_KC, NW1_NT);
    pack_b_kernel<<<CDIV(NW2_KC * NW2_NT * 32, 256), 256, 0, stream>>>(
        nw2 + (size_t)id * 256 * DD, pNw2[id], 256, DD, NW2_KC, NW2_NT);
  }

  hipMemsetAsync(pooled, 0, (size_t)AA * GG * DD * 4, stream);

  for (int adj = 0; adj < AA; ++adj) {
    // state0 = [coord | relu(x @ lin0_w + b)]  (re-derived per adj: cheap, stateless)
    lin0_kernel<<<CDIV(NN * DD, 256), 256, 0, stream>>>(x, lin0_w, lin0_b, feats, featsBf);
    copy_coords_kernel<<<CDIV(NN * 3, 256), 256, 0, stream>>>(coord, coordsW);

    const long long* rowp = eidx + (size_t)adj * 2 * EE;
    const long long* colp = rowp + EE;

    for (int l = 0; l < LL; ++l) {
      int id = adj * LL + l;
      hipMemsetAsync(mi, 0, (size_t)NN * DD * 4, stream);
      edge_kernel<<<EE / BE, 256, 0, stream>>>(
          rowp, colp, featsBf, coordsW,
          pEw1[id], rEw1[id], eb1 + (size_t)id * EH1,
          pEw2[id], eb2 + (size_t)id * DD, mi);
      node_kernel<<<NN / BNODE, 256, 0, stream>>>(
          feats, featsBf, mi, coordsW,
          pNw1[id], nb1 + (size_t)id * 256,
          pNw2[id], nb2 + (size_t)id * DD);
    }
    pool_kernel<<<CDIV(NN * DD, 256), 256, 0, stream>>>(
        feats, batch, pooled + (size_t)adj * GG * DD);
  }

  final_kernel<<<1, 32, 0, stream>>>(pooled, l1w, l1b, l2w, l2b, (float*)d_out);
}
